// HiResPrecipNet_up_CNN_GNN_70248485093878
// MI455X (gfx1250) — matrix operand order repaired and checked by reading, not compiled, not run on hardware
//
#include <hip/hip_runtime.h>
#include <math.h>

typedef __attribute__((ext_vector_type(16))) _Float16 v16h;
typedef __attribute__((ext_vector_type(8)))  float    v8f;

#define TPB 256

// ---------------------------------------------------------------- utilities
__global__ void k_fill(float* __restrict__ p, float v, int n) {
  int i = blockIdx.x * blockDim.x + threadIdx.x;
  if (i < n) p[i] = v;
}

__device__ __forceinline__ void atomicMaxF(float* a, float v) {
  if (v >= 0.0f) atomicMax((int*)a, __float_as_int(v));
  else           atomicMin((unsigned int*)a, __float_as_uint(v));
}

// ---------------------------------------------------------------- conv1d
// in: [B, Cin, L]; tin=1: input is x_low laid out [L, B] with Cin==1
__global__ void k_conv1d(const float* __restrict__ in, const float* __restrict__ w,
                         const float* __restrict__ b, float* __restrict__ out,
                         int B, int Cin, int Cout, int L, int Ksz, int pad,
                         int act_tanh, int tin)
{
  int idx = blockIdx.x * blockDim.x + threadIdx.x;
  int total = B * Cout * L;
  if (idx >= total) return;
  int x  = idx % L;
  int co = (idx / L) % Cout;
  int bb = idx / (L * Cout);
  float acc = b[co];
  for (int ci = 0; ci < Cin; ++ci) {
    const float* wp = w + (co * Cin + ci) * Ksz;
    for (int k = 0; k < Ksz; ++k) {
      int xx = x + k - pad;
      if (xx < 0 || xx >= L) continue;
      float v = tin ? in[xx * B + bb] : in[(bb * Cin + ci) * L + xx];
      acc += v * wp[k];
    }
  }
  out[idx] = act_tanh ? tanhf(acc) : acc;
}

// ---------------------------------------------------------------- depthwise conv2d
// kernel 3x5, pad (1,1): outH = H, outW = W-2.
// fromT3=1: input element (n, c, iy, ix) is read from the upscaler output
// T3 [625, 25, 1000] with b2 = c*125 + iy*25 + ix, s = n%25, nn = n/25.
__global__ void k_dwconv2d(const float* __restrict__ in, const float* __restrict__ w,
                           const float* __restrict__ b, float* __restrict__ out,
                           int N, int C, int H, int W, int fromT3)
{
  int outW = W - 2;
  int idx = blockIdx.x * blockDim.x + threadIdx.x;
  int total = N * C * H * outW;
  if (idx >= total) return;
  int ox = idx % outW;
  int oy = (idx / outW) % H;
  int c  = (idx / (outW * H)) % C;
  int n  = idx / (outW * H * C);
  float acc = b[c];
  for (int ky = 0; ky < 3; ++ky) {
    int iy = oy + ky - 1;
    if (iy < 0 || iy >= H) continue;
    for (int kx = 0; kx < 5; ++kx) {
      int ix = ox + kx - 1;
      if (ix < 0 || ix >= W) continue;
      float v;
      if (fromT3) {
        int b2 = c * 125 + iy * 25 + ix;
        v = in[b2 * 25000 + (n % 25) * 1000 + (n / 25)];
      } else {
        v = in[((n * C + c) * H + iy) * W + ix];
      }
      acc += v * w[(c * 3 + ky) * 5 + kx];
    }
  }
  out[idx] = acc;
}

// ---------------------------------------------------------------- batch norm (training stats)
__global__ void k_bn_stats(const float* __restrict__ x, float* __restrict__ stats,
                           int total, int C, int HW)
{
  extern __shared__ float sh[];
  for (int i = threadIdx.x; i < 2 * C; i += blockDim.x) sh[i] = 0.0f;
  __syncthreads();
  for (int i = blockIdx.x * blockDim.x + threadIdx.x; i < total;
       i += gridDim.x * blockDim.x) {
    int c = (i / HW) % C;
    float v = x[i];
    atomicAdd(&sh[c], v);
    atomicAdd(&sh[C + c], v * v);
  }
  __syncthreads();
  for (int i = threadIdx.x; i < 2 * C; i += blockDim.x) atomicAdd(&stats[i], sh[i]);
}

__global__ void k_bn_apply(float* __restrict__ x, const float* __restrict__ stats,
                           const float* __restrict__ g, const float* __restrict__ be,
                           int total, int C, int HW, float inv_n, int relu)
{
  int i = blockIdx.x * blockDim.x + threadIdx.x;
  if (i >= total) return;
  int c = (i / HW) % C;
  float mean = stats[c] * inv_n;
  float var  = stats[C + c] * inv_n - mean * mean;
  float y = (x[i] - mean) * rsqrtf(var + 1e-5f) * g[c] + be[c];
  if (relu) y = fmaxf(y, 0.0f);
  x[i] = y;
}

// ---------------------------------------------------------------- maxpool 2x2 + flatten
__global__ void k_pool_flat(const float* __restrict__ in, float* __restrict__ out, int N)
{
  int idx = blockIdx.x * blockDim.x + threadIdx.x;
  if (idx >= N * 90) return;
  int j = idx % 9;
  int i = (idx / 9) % 2;
  int c = (idx / 18) % 5;
  int n = idx / 90;
  float m = -3.0e38f;
  for (int di = 0; di < 2; ++di)
    for (int dj = 0; dj < 2; ++dj) {
      float v = in[((n * 5 + c) * 5 + (2 * i + di)) * 19 + (2 * j + dj)];
      m = fmaxf(m, v);
    }
  out[idx] = m;
}

// ---------------------------------------------------------------- WMMA GEMM
// out[M,N] = relu?( A[M,K] @ W[K,N] + bias[N] ). f32 data, f16 WMMA, f32 accum.
// Block: 256 threads = 8 waves; block tile 128 rows x 16 cols; one wave per 16x16.
// A tile: clamped UNCONDITIONAL loads (no value select -> no load predication);
//         out-of-range-K garbage is annihilated by exact zeros on the B side.
// B tile: staged via CDNA5 async tensor path (global_load_async_to_lds_b32);
//         out-of-range-K entries zeroed with a branch-free LDS-side select.
__global__ void __launch_bounds__(256)
k_gemm_wmma(const float* __restrict__ A, const float* __restrict__ W,
            const float* __restrict__ bias, float* __restrict__ out,
            int M, int K, int N, int relu)
{
  __shared__ _Float16 sA[128 * 34];     // [row_local][k], padded stride 34 (even)
  __shared__ float    sB[32 * 16];      // [k][n] raw f32, async-filled

  const int t      = threadIdx.x;
  const int lane   = t & 31;
  const int wv     = t >> 5;
  const int l15    = lane & 15;
  const int hiHalf = lane >> 4;          // 0: lanes 0-15, 1: lanes 16-31
  const int row0   = blockIdx.x * 128;
  const int col0   = blockIdx.y * 16;
  const int kbA    = hiHalf ? 8 : 0;     // A K-base per ISA 16-bit A layout
  const int kbB    = hiHalf ? 8 : 0;     // B K-base, same table order as A

  v8f c = {};
  for (int kk = 0; kk < K; kk += 32) {
    // ---- stage A tile: 128 x 32, coalesced, branch-free (clamped addresses,
    //      no zero-select: garbage K-lanes hit zeroed B entries).
#pragma unroll
    for (int it = 0; it < 16; ++it) {
      int i    = t + it * 256;          // 0..4095
      int r    = i >> 5;                // local row
      int kc   = i & 31;                // local k
      int grow = row0 + r;
      int gk   = kk + kc;
      int sr   = grow < M ? grow : 0;   // clamped (always-legal) address
      int sk   = gk < K ? gk : 0;
      sA[r * 34 + kc] = (_Float16)A[sr * K + sk];
    }
    // ---- stage B tile: 32 x 16 via async load-to-LDS (clamped-safe addresses)
#pragma unroll
    for (int it = 0; it < 2; ++it) {
      int i  = t + it * 256;            // 0..511
      int kq = i >> 4;                  // local k
      int nq = i & 15;                  // local n
      int gk = kk + kq;
      int gn = col0 + nq;
      int sk = gk < K ? gk : 0;
      int sn = gn < N ? gn : 0;
      const float* gp = W + (size_t)sk * N + sn;
      unsigned lds = (unsigned)(size_t)(&sB[i]);
      asm volatile("global_load_async_to_lds_b32 %0, %1, off"
                   :: "v"(lds), "v"((unsigned long long)(size_t)gp)
                   : "memory");
    }
    asm volatile("s_wait_asynccnt 0x0" ::: "memory");
    __syncthreads();

    // ---- build fragments from LDS
    v16h a, b;
    const _Float16* ap = &sA[(wv * 16 + l15) * 34 + kbA];
#pragma unroll
    for (int v = 0; v < 8; ++v) {
      int kofs = (v < 4) ? (2 * v) : (16 + 2 * (v - 4));
      a[2 * v]     = ap[kofs];
      a[2 * v + 1] = ap[kofs + 1];
    }
#pragma unroll
    for (int hi = 0; hi < 16; ++hi) {
      const int kb = kbB + hi + (hi >= 8 ? 8 : 0);
      float bv = sB[kb * 16 + l15];
      bv = (kk + kb < K) ? bv : 0.0f;   // exact zeros for OOB-K (branch-free)
      b[hi] = (_Float16)bv;
    }
    c = __builtin_amdgcn_wmma_f32_16x16x32_f16(false, a, false, b, (short)0, c,
                                               false, false);
    __syncthreads();
  }

  // ---- epilogue: hoisted bias, fast path for full tiles ----
  const int col  = col0 + l15;
  const float bcol = bias[col < N ? col : 0];   // clamped, loaded once
  const int rowBase = row0 + wv * 16 + (hiHalf ? 8 : 0);
  if (col < N) {
    if (rowBase + 7 < M) {                      // full tile: 8 plain stores
#pragma unroll
      for (int r = 0; r < 8; ++r) {
        float v = c[r] + bcol;
        if (relu) v = fmaxf(v, 0.0f);
        out[(rowBase + r) * N + col] = v;
      }
    } else {                                    // ragged M edge
#pragma unroll
      for (int r = 0; r < 8; ++r) {
        if (rowBase + r < M) {
          float v = c[r] + bcol;
          if (relu) v = fmaxf(v, 0.0f);
          out[(rowBase + r) * N + col] = v;
        }
      }
    }
  }
}

// ---------------------------------------------------------------- GATv2 edge kernels
__global__ void k_edge_score(const float* __restrict__ xl, const float* __restrict__ xr,
                             const int* __restrict__ src, const int* __restrict__ dst,
                             const float* __restrict__ att, float* __restrict__ score,
                             float* __restrict__ smax, int E, int Hh, int D)
{
  int e = blockIdx.x * blockDim.x + threadIdx.x;
  if (e >= E) return;
  int s = src[e], d = dst[e];
  const int HD = Hh * D;
  __builtin_prefetch(xl + s * HD, 0, 1);
  __builtin_prefetch(xr + d * HD, 0, 1);
  for (int h = 0; h < Hh; ++h) {
    const float* pl = xl + s * HD + h * D;
    const float* pr = xr + d * HD + h * D;
    const float* pa = att + h * D;
    float sc = 0.0f;
    for (int j = 0; j < D; ++j) {
      float g = pl[j] + pr[j];
      g = (g > 0.0f) ? g : 0.2f * g;          // leaky_relu(0.2)
      sc += g * pa[j];
    }
    score[e * Hh + h] = sc;
    atomicMaxF(&smax[d * Hh + h], sc);
  }
}

__global__ void k_edge_exp(const int* __restrict__ dst, float* __restrict__ score,
                           const float* __restrict__ smax, float* __restrict__ den,
                           int E, int Hh)
{
  int i = blockIdx.x * blockDim.x + threadIdx.x;
  if (i >= E * Hh) return;
  int e = i / Hh, h = i % Hh;
  int d = dst[e];
  float a = expf(score[i] - smax[d * Hh + h]);
  score[i] = a;
  atomicAdd(&den[d * Hh + h], a);
}

__global__ void k_edge_scatter(const float* __restrict__ xl, const int* __restrict__ src,
                               const int* __restrict__ dst, const float* __restrict__ score,
                               const float* __restrict__ den, float* __restrict__ sacc,
                               float* __restrict__ deg, int E, int Hh, int D)
{
  int e = blockIdx.x * blockDim.x + threadIdx.x;
  if (e >= E) return;
  int s = src[e], d = dst[e];
  const int HD = Hh * D;
  __builtin_prefetch(xl + s * HD, 0, 1);
  for (int h = 0; h < Hh; ++h) {
    float alpha = score[e * Hh + h] / (den[d * Hh + h] + 1e-16f);
    const float* pl = xl + s * HD + h * D;
    float* po = sacc + d * HD + h * D;
    for (int j = 0; j < D; ++j) atomicAdd(&po[j], alpha * pl[j]);
  }
  atomicAdd(&deg[d], 1.0f);
}

__global__ void k_gat_final(const float* __restrict__ sacc, const float* __restrict__ deg,
                            const float* __restrict__ bias, float* __restrict__ out,
                            int N, int F, int relu)
{
  int i = blockIdx.x * blockDim.x + threadIdx.x;
  if (i >= N * F) return;
  int n = i / F, f = i % F;
  float v = sacc[i] / fmaxf(deg[n], 1.0f) + bias[f];
  if (relu) v = fmaxf(v, 0.0f);
  out[i] = v;
}

// ---------------------------------------------------------------- prediction MLP
__global__ void k_pred(const float* __restrict__ h, const float* __restrict__ W1,
                       const float* __restrict__ b1, const float* __restrict__ W2,
                       const float* __restrict__ b2, const float* __restrict__ W3,
                       const float* __restrict__ b3, float* __restrict__ out, int N)
{
  __shared__ float sW1[256], sb1[16], sW2[512], sb2[32], sW3[32], sb3[1];
  int t = threadIdx.x;
  for (int i = t; i < 256; i += blockDim.x) sW1[i] = W1[i];
  for (int i = t; i < 512; i += blockDim.x) sW2[i] = W2[i];
  if (t < 32) { sW3[t] = W3[t]; sb2[t] = b2[t]; }
  if (t < 16) sb1[t] = b1[t];
  if (t == 0) sb3[0] = b3[0];
  __syncthreads();
  int n = blockIdx.x * blockDim.x + t;
  if (n >= N) return;
  float x[16], y1[16], y2[32];
  for (int i = 0; i < 16; ++i) x[i] = h[n * 16 + i];
  for (int j = 0; j < 16; ++j) {
    float a = sb1[j];
    for (int i = 0; i < 16; ++i) a += x[i] * sW1[i * 16 + j];
    y1[j] = fmaxf(a, 0.0f);
  }
  for (int j = 0; j < 32; ++j) {
    float a = sb2[j];
    for (int i = 0; i < 16; ++i) a += y1[i] * sW2[i * 32 + j];
    y2[j] = fmaxf(a, 0.0f);
  }
  float o = sb3[0];
  for (int i = 0; i < 32; ++i) o += y2[i] * sW3[i];
  out[n] = o;
}

// ================================================================ host driver
#define FIN(i) ((const float*)d_in[(i)])
#define IIN(i) ((const int*)d_in[(i)])

extern "C" void kernel_launch(void* const* d_in, const int* in_sizes, int n_in,
                              void* d_out, int out_size, void* d_ws, size_t ws_size,
                              hipStream_t stream)
{
  (void)in_sizes; (void)n_in; (void)out_size; (void)ws_size;
  float* ws = (float*)d_ws;

  auto g1d = [](int total) { return dim3((total + TPB - 1) / TPB); };

  // ---------------- workspace layout (floats) ----------------
  float* T1 = ws + 0;          // [625, 64, 1000]  40.0M
  float* T2 = ws + 40000000;   // [625, 32, 1000]  20.0M
  float* T3 = ws + 60000000;   // [625, 25, 1000]  15.625M
  float* EA = ws + 0;          // [25000,5,5,23]   reuses T1 region
  float* EB = ws + 40000000;   // [25000,5,5,21]   reuses T2 region
  float* EC = ws + 60000000;   // [25000,5,5,19]   reuses T3 region
  float* H0   = ws + 0;          // node feature ping (<=4.0M)
  float* H1   = ws + 4000000;    // node feature pong
  float* XL   = ws + 8000000;    // source transform (<=3.2M)
  float* XR   = ws + 12000000;   // target transform
  float* SC   = ws + 16000000;   // edge scores (<=2.4M)
  float* SMAX = ws + 18600000;   // per-(dst,head) max
  float* DEN  = ws + 18900000;   // per-(dst,head) softmax denom
  float* SACC = ws + 19200000;   // aggregation accumulator (<=3.2M)
  float* DEG  = ws + 23300000;   // per-dst degree
  float* STAT = ws + 23500000;   // BN sum/sumsq (<=128)

  auto gemm = [&](const float* A, const float* W, const float* bias, float* out,
                  int M, int K, int N, int relu) {
    dim3 grid((M + 127) / 128, (N + 15) / 16);
    k_gemm_wmma<<<grid, 256, 0, stream>>>(A, W, bias, out, M, K, N, relu);
  };

  auto bn = [&](float* x, const float* g, const float* be, int N, int C, int HW) {
    int total = N * C * HW;
    k_fill<<<1, 256, 0, stream>>>(STAT, 0.0f, 2 * C);
    int blocks = (total + TPB - 1) / TPB;
    if (blocks > 2048) blocks = 2048;
    k_bn_stats<<<blocks, TPB, 2 * C * sizeof(float), stream>>>(x, STAT, total, C, HW);
    k_bn_apply<<<g1d(total), TPB, 0, stream>>>(x, STAT, g, be, total, C, HW,
                                               1.0f / (float)(N * HW), 1);
  };

  auto gat = [&](const float* xs, int Ns, int dS, const float* xd, int Nd, int dD,
                 int pbase, int Hh, int D, const int* src, const int* dst, int E,
                 float* out, int relu_out) {
    const float *Wl = FIN(pbase),     *bl = FIN(pbase + 1);
    const float *Wr = FIN(pbase + 2), *br = FIN(pbase + 3);
    const float *att = FIN(pbase + 4), *bias = FIN(pbase + 5);
    int HD = Hh * D;
    gemm(xs, Wl, bl, XL, Ns, dS, HD, 0);
    gemm(xd, Wr, br, XR, Nd, dD, HD, 0);
    k_fill<<<g1d(Nd * Hh), TPB, 0, stream>>>(SMAX, -3.0e38f, Nd * Hh);
    k_fill<<<g1d(Nd * Hh), TPB, 0, stream>>>(DEN, 0.0f, Nd * Hh);
    k_fill<<<g1d(Nd * HD), TPB, 0, stream>>>(SACC, 0.0f, Nd * HD);
    k_fill<<<g1d(Nd), TPB, 0, stream>>>(DEG, 0.0f, Nd);
    k_edge_score<<<g1d(E), TPB, 0, stream>>>(XL, XR, src, dst, att, SC, SMAX, E, Hh, D);
    k_edge_exp<<<g1d(E * Hh), TPB, 0, stream>>>(dst, SC, SMAX, DEN, E, Hh);
    k_edge_scatter<<<g1d(E), TPB, 0, stream>>>(XL, src, dst, SC, DEN, SACC, DEG, E, Hh, D);
    k_gat_final<<<g1d(Nd * HD), TPB, 0, stream>>>(SACC, DEG, bias, out, Nd, HD, relu_out);
  };

  // ---------------- input pointer map (setup_inputs order, flattened) ----------
  const float* x_low = FIN(0);
  const float* z_std = FIN(1);

  // ---------------- stage 1: sub-pixel upscaler --------------------------------
  k_conv1d<<<g1d(625 * 64 * 1000), TPB, 0, stream>>>(x_low, FIN(2), FIN(3), T1,
                                                     625, 1, 64, 1000, 5, 2, 1, 1);
  k_conv1d<<<g1d(625 * 32 * 1000), TPB, 0, stream>>>(T1, FIN(4), FIN(5), T2,
                                                     625, 64, 32, 1000, 3, 1, 1, 0);
  k_conv1d<<<g1d(625 * 25 * 1000), TPB, 0, stream>>>(T2, FIN(6), FIN(7), T3,
                                                     625, 32, 25, 1000, 3, 1, 0, 0);

  // ---------------- stage 2: encoder -------------------------------------------
  k_dwconv2d<<<g1d(25000 * 5 * 5 * 23), TPB, 0, stream>>>(T3, FIN(8), FIN(9), EA,
                                                          25000, 5, 5, 25, 1);
  bn(EA, FIN(10), FIN(11), 25000, 5, 5 * 23);
  k_dwconv2d<<<g1d(25000 * 5 * 5 * 21), TPB, 0, stream>>>(EA, FIN(12), FIN(13), EB,
                                                          25000, 5, 5, 23, 0);
  bn(EB, FIN(14), FIN(15), 25000, 5, 5 * 21);
  k_dwconv2d<<<g1d(25000 * 5 * 5 * 19), TPB, 0, stream>>>(EB, FIN(16), FIN(17), EC,
                                                          25000, 5, 5, 21, 0);
  bn(EC, FIN(18), FIN(19), 25000, 5, 5 * 19);
  k_pool_flat<<<g1d(25000 * 90), TPB, 0, stream>>>(EC, H0, 25000);

  const int* elS = IIN(108); const int* elD = IIN(109);
  const int* ebS = IIN(110); const int* ebD = IIN(111);
  const int* ehS = IIN(112); const int* ehD = IIN(113);
  const int heads[5] = {2, 2, 2, 2, 1};

  // ---------------- stage 3: processor_low -------------------------------------
  float* h = H0; float* o = H1;
  int din = 90;
  for (int i = 0; i < 5; ++i) {
    int Hh = heads[i]; int outF = Hh * 32;
    gat(h, 25000, din, h, 25000, din, 20 + 6 * i, Hh, 32, elS, elD, 400000, o,
        (i == 4) ? 1 : 0);
    if (i < 4) bn(o, FIN(50 + 2 * i), FIN(51 + 2 * i), 25000, outF, 1);
    float* t = h; h = o; o = t; din = outF;
  }

  // ---------------- stage 4: bipartite downscaler ------------------------------
  gat(h, 25000, 32, z_std, 100000, 1, 58, 1, 16, ebS, ebD, 600000, o, 0);
  { float* t = h; h = o; o = t; }
  din = 16;

  // ---------------- stage 5: processor high ------------------------------------
  for (int i = 0; i < 5; ++i) {
    int Hh = heads[i]; int outF = Hh * 16;
    gat(h, 100000, din, h, 100000, din, 64 + 6 * i, Hh, 16, ehS, ehD, 1200000, o,
        (i == 4) ? 1 : 0);
    if (i < 4) bn(o, FIN(94 + 2 * i), FIN(95 + 2 * i), 100000, outF, 1);
    float* t = h; h = o; o = t; din = outF;
  }

  // ---------------- stage 6: prediction head -----------------------------------
  k_pred<<<g1d(100000), TPB, 0, stream>>>(h, FIN(102), FIN(103), FIN(104), FIN(105),
                                          FIN(106), FIN(107), (float*)d_out, 100000);
}
